// VoxelNet_15006615733709
// MI455X (gfx1250) — matrix operand order repaired and checked by reading, not compiled, not run on hardware
//
#include <hip/hip_runtime.h>
#include <hip/hip_bf16.h>

typedef __attribute__((ext_vector_type(16))) _Float16 v16h;
typedef __attribute__((ext_vector_type(8)))  float    v8f;

#define NXg 176
#define NYg 200
#define NZg 10
#define NCELL (NXg*NYg*NZg)
#define PV 32
#define MAXV 20000
#define BATCH 2
#define NPTS 100000

// ---------------------------------------------------------------- voxelize
__device__ inline int point_vid(float x, float y, float z) {
    float fx = floorf((x - 0.0f)  / 0.4f);
    float fy = floorf((y + 40.0f) / 0.4f);
    float fz = floorf((z + 3.0f)  / 0.4f);
    int ix = (int)fx, iy = (int)fy, iz = (int)fz;
    if (ix < 0 || iy < 0 || iz < 0 || ix >= NXg || iy >= NYg || iz >= NZg) return -1;
    return (ix * NYg + iy) * NZg + iz;
}

__global__ void k_count(const float* __restrict__ pts, int* __restrict__ cellcount) {
    int idx = blockIdx.x * 256 + threadIdx.x;
    if (idx >= BATCH * NPTS) return;
    int b = idx / NPTS;
    const float* p = pts + (size_t)idx * 4;
    int vid = point_vid(p[0], p[1], p[2]);
    if (vid >= 0) atomicAdd(&cellcount[b * NCELL + vid], 1);
}

// One block (1024 thr) per batch: exclusive prefix of occupied flags -> ordinal
__global__ void k_scan(const int* __restrict__ cellcount, int* __restrict__ ordinal) {
    int b = blockIdx.x;
    __shared__ int buf[1024];
    __shared__ int running;
    if (threadIdx.x == 0) running = 0;
    __syncthreads();
    for (int base = 0; base < NCELL; base += 1024) {
        int i = base + threadIdx.x;
        int flag = (i < NCELL && cellcount[b * NCELL + i] > 0) ? 1 : 0;
        buf[threadIdx.x] = flag;
        __syncthreads();
        for (int off = 1; off < 1024; off <<= 1) {
            int v = buf[threadIdx.x];
            int a = (threadIdx.x >= off) ? buf[threadIdx.x - off] : 0;
            __syncthreads();
            buf[threadIdx.x] = v + a;
            __syncthreads();
        }
        int incl = buf[threadIdx.x];
        if (i < NCELL) ordinal[b * NCELL + i] = running + incl - flag;
        __syncthreads();
        if (threadIdx.x == 1023) running += incl;
        __syncthreads();
    }
}

__global__ void k_fill(const float* __restrict__ pts, const int* __restrict__ cellcount,
                       const int* __restrict__ ordinal, int* __restrict__ slot,
                       int* __restrict__ vvv, float* __restrict__ vox) {
    int idx = blockIdx.x * 256 + threadIdx.x;
    if (idx >= BATCH * NPTS) return;
    int b = idx / NPTS;
    const float* p = pts + (size_t)idx * 4;
    int vid = point_vid(p[0], p[1], p[2]);
    if (vid < 0) return;
    int o = ordinal[b * NCELL + vid];
    if (o >= MAXV) return;
    int gv = b * MAXV + o;
    int rank = atomicAdd(&slot[gv], 1);
    if (rank == 0) vvv[gv] = vid;
    if (rank < PV) {
        float* vp = vox + ((size_t)gv * PV + rank) * 4;
        vp[0] = p[0]; vp[1] = p[1]; vp[2] = p[2]; vp[3] = p[3];
    }
}

// one wave per voxel: mean-centered 7-channel features, zeroed on pad slots
__global__ void k_feat(const float* __restrict__ vox, const int* __restrict__ slot,
                       float* __restrict__ feat) {
    int t = threadIdx.x;
    int gv = blockIdx.x * 4 + (t >> 5);
    int lane = t & 31;
    int cnt = min(slot[gv], PV);
    const float* vp = vox + (size_t)gv * PV * 4;
    float x = vp[lane * 4 + 0], y = vp[lane * 4 + 1];
    float z = vp[lane * 4 + 2], r = vp[lane * 4 + 3];
    float sx = x, sy = y, sz = z;
    for (int o = 16; o >= 1; o >>= 1) {
        sx += __shfl_xor(sx, o);
        sy += __shfl_xor(sy, o);
        sz += __shfl_xor(sz, o);
    }
    float dn = (float)max(cnt, 1);
    float mx = sx / dn, my = sy / dn, mz = sz / dn;
    float m = (lane < cnt) ? 1.f : 0.f;
    float* fp = feat + ((size_t)gv * PV + lane) * 7;
    fp[0] = x * m; fp[1] = y * m; fp[2] = z * m; fp[3] = r * m;
    fp[4] = (x - mx) * m; fp[5] = (y - my) * m; fp[6] = (z - mz) * m;
}

// ---------------------------------------------------------------- weight packing
// WMMA A-fragment swizzle offset within a 32-wide k-tile for (hi, j)
__device__ __host__ inline int koffA(int hi, int j) {
    return j + 8 * hi + ((j >= 8) ? 8 : 0);
}

// Conv weights -> fragment-major A layout: out[o*576 + kt*32 + hi*16 + j] = w[o][kt*32+hi*16+j]
__global__ void k_pack_convA(const float* __restrict__ w, _Float16* __restrict__ out, int n) {
    int idx = blockIdx.x * 256 + threadIdx.x;
    if (idx >= n) return;
    int o = idx / 576, t = idx - o * 576;
    int kt = t >> 5, rem = t & 31;
    int hi = rem >> 4, j = rem & 15;
    int k = kt * 32 + rem;
    out[idx] = (_Float16)w[(size_t)o * 576 + k];
}

// PFN weights [K][C] -> fragment-major B layout:
// out[(((kt*2+hi)*C)+n)*16 + j] = (k<K ? w[k][n] : 0), k = kt*32 + 16*hi + j
__global__ void k_pack_wB(const float* __restrict__ w, _Float16* __restrict__ out,
                          int K, int Kpad, int C) {
    int idx = blockIdx.x * 256 + threadIdx.x;
    if (idx >= Kpad * C) return;
    int j = idx & 15;
    int tmp = idx >> 4;
    int n = tmp % C;
    int g = tmp / C;
    int kt = g >> 1, hi = g & 1;
    int k = kt * 32 + 16 * hi + j;
    out[idx] = (_Float16)((k < K) ? w[(size_t)k * C + n] : 0.f);
}

__global__ void k_f32_to_f16(const float* __restrict__ s, _Float16* __restrict__ d, int n) {
    int idx = blockIdx.x * 256 + threadIdx.x;
    if (idx < n) d[idx] = (_Float16)s[idx];
}

__global__ void k_finalize_bn(const float* __restrict__ cs, const float* __restrict__ cq,
                              const float* __restrict__ g, const float* __restrict__ b,
                              float* __restrict__ scale, float* __restrict__ bias,
                              int C, float N, float eps) {
    int c = threadIdx.x;
    if (c >= C) return;
    float mean = cs[c] / N;
    float var = cq[c] / N - mean * mean;
    float rs = rsqrtf(var + eps);
    scale[c] = g[c] * rs;
    bias[c] = b[c] - mean * g[c] * rs;
}

// ---------------------------------------------------------------- PFN GEMM (WMMA)
// block = 128 thr (4 waves) handles 64 rows = 2 voxels. Rows staged to LDS in
// A-fragment-major order so fragment reads are aligned 32B ds_load_b128 pairs.
// mode 0: column sum/sumsq (BN stats over ALL rows, pads included)
// mode 1: BN+relu, masked per-voxel max -> vmaxout[B*MAXV][C]
// mode 2: BN+relu, masked per-voxel max -> atomicMax scatter into dense BEV
__global__ void k_pfn(const float* __restrict__ feat, const float* __restrict__ vmaxprev,
                      int Cprev, int Kpad, const _Float16* __restrict__ wqP, int C, int mode,
                      const int* __restrict__ slotcnt, const int* __restrict__ vvv,
                      float* __restrict__ colsum, float* __restrict__ colsumsq,
                      const float* __restrict__ bnscale, const float* __restrict__ bnbias,
                      float* __restrict__ vmaxout, float* __restrict__ dense) {
    __shared__ __align__(32) _Float16 rowsP[64][96];  // stride 192B (32B multiple)
    __shared__ float red[4][64];
    __shared__ int scnt[2], svv[2];
    int tid = threadIdx.x, lane = tid & 31, w = tid >> 5;
    int hi = lane >> 4, ln = lane & 15;
    size_t row0 = (size_t)blockIdx.x * 64;
    int gv0 = blockIdx.x * 2;
    if (tid < 2) { scnt[tid] = min(slotcnt[gv0 + tid], PV); svv[tid] = vvv[gv0 + tid]; }
    // stage rows in plain k order: rowsP[r][k]
    for (int idx = tid; idx < 64 * Kpad; idx += 128) {
        int r = idx / Kpad, t = idx - r * Kpad;
        int kt = t >> 5, rem = t & 31;
        int h2 = rem >> 4, j = rem & 15;
        int k = kt * 32 + rem;
        size_t row = row0 + r;
        size_t gv = row >> 5;
        int p = (int)(row & 31);
        float v = 0.f;
        if (k < 7)              v = feat[(gv * PV + p) * 7 + k];
        else if (k < 7 + Cprev) v = vmaxprev[gv * Cprev + (k - 7)];
        rowsP[r][t] = (_Float16)v;
    }
    __syncthreads();
    int NT = C >> 4;
    for (int nt = 0; nt < NT; nt++) {
        int n0 = nt << 4;
        v8f acc = {0.f, 0.f, 0.f, 0.f, 0.f, 0.f, 0.f, 0.f};
        for (int k0 = 0; k0 < Kpad; k0 += 32) {
            int kt = k0 >> 5;
            v16h a  = *(const v16h*)&rowsP[w * 16 + ln][kt * 32 + hi * 16];
            v16h bf = *(const v16h*)(wqP + ((size_t)((kt * 2 + hi) * C + n0 + ln)) * 16);
            acc = __builtin_amdgcn_wmma_f32_16x16x32_f16(false, a, false, bf,
                                                         (short)0, acc, false, false);
        }
        if (mode == 0) {
            float s = 0.f, q = 0.f;
#pragma unroll
            for (int r = 0; r < 8; r++) { float v = acc[r]; s += v; q += v * v; }
            s += __shfl_xor(s, 16);
            q += __shfl_xor(q, 16);
            if (lane < 16) {
                atomicAdd(&colsum[n0 + lane], s);
                atomicAdd(&colsumsq[n0 + lane], q);
            }
        } else {
            int cc = n0 + ln;
            float sc = bnscale[cc], bi = bnbias[cc];
            int cnt = scnt[w >> 1];
            float mx = 0.f;
#pragma unroll
            for (int r = 0; r < 8; r++) {
                int p = ((w & 1) << 4) + r + 8 * hi;
                float v = fmaxf(0.f, acc[r] * sc + bi);
                if (p < cnt) mx = fmaxf(mx, v);
            }
            mx = fmaxf(mx, __shfl_xor(mx, 16));
            if (lane < 16) red[w][cc] = mx;
        }
    }
    if (mode != 0) {
        __syncthreads();
        for (int idx = tid; idx < 2 * C; idx += 128) {
            int vx = idx / C, ch = idx - vx * C;
            float v = fmaxf(red[2 * vx + 0][ch], red[2 * vx + 1][ch]);
            size_t gv = (size_t)gv0 + vx;
            if (mode == 1) {
                vmaxout[gv * C + ch] = v;
            } else {
                int vid = svv[vx];
                int cx = vid / (NYg * NZg);
                int cy = (vid / NZg) % NYg;
                int bb = (int)(gv / MAXV);
                size_t di = (((size_t)(bb * 64 + ch)) * NYg + cy) * NXg + cx;
                atomicMax((int*)&dense[di], __float_as_int(v));
            }
        }
    }
}

// ---------------------------------------------------------------- conv 3x3 (implicit GEMM, WMMA)
// grid (ceil(W/32), H, B), 256 thr (8 waves): 64 outC x 32 pixels per block.
// Patch staged px-major (48-half stride) so B fragments are aligned v16h LDS loads;
// weights pre-packed fragment-major so A fragments are aligned v16h global loads.
__global__ void k_conv(const _Float16* __restrict__ xh, const _Float16* __restrict__ wpA,
                       const float* __restrict__ bias, float* __restrict__ yout) {
    __shared__ __align__(32) _Float16 patch[32][48];  // stride 96B (32B multiple)
    int b = blockIdx.z, yrow = blockIdx.y, x0 = blockIdx.x * 32;
    int tid = threadIdx.x, lane = tid & 31, w = tid >> 5;
    int o0 = (w & 3) * 16, pt = w >> 2;
    int hi = lane >> 4, ln = lane & 15;
    const _Float16* aBase = wpA + (size_t)(o0 + ln) * 576 + hi * 16;
    v8f acc = {0.f, 0.f, 0.f, 0.f, 0.f, 0.f, 0.f, 0.f};
    for (int kt = 0; kt < 18; kt++) {
        int k0 = kt * 32;
        __syncthreads();
        for (int idx = tid; idx < 1024; idx += 256) {
            int px = idx >> 5, kk = idx & 31;
            int k = k0 + kk;
            int i = k / 9, rem = k - i * 9;
            int ky = rem / 3, kx = rem - ky * 3;
            int yy = yrow + ky - 1, xx = x0 + px + kx - 1;
            _Float16 v = (_Float16)0.f;
            if (yy >= 0 && yy < NYg && xx >= 0 && xx < NXg)
                v = xh[(((size_t)(b * 64 + i)) * NYg + yy) * NXg + xx];
            patch[px][kk] = v;
        }
        if (kt + 1 < 18)  // prefetch next k-tile's A fragment (global_prefetch_b8)
            __builtin_prefetch(aBase + (size_t)(kt + 1) * 32, 0, 1);
        __syncthreads();
        v16h a  = *(const v16h*)(aBase + (size_t)kt * 32);
        v16h bf = *(const v16h*)&patch[pt * 16 + ln][16 * hi];
        acc = __builtin_amdgcn_wmma_f32_16x16x32_f16(false, a, false, bf,
                                                     (short)0, acc, false, false);
    }
    int xx = x0 + pt * 16 + ln;
    if (xx < NXg) {
#pragma unroll
        for (int r = 0; r < 8; r++) {
            int o = o0 + r + 8 * hi;
            yout[(((size_t)(b * 64 + o)) * NYg + yrow) * NXg + xx] = acc[r] + bias[o];
        }
    }
}

// one block per channel: deterministic per-channel BN stats over B*H*W
__global__ void k_conv_stats(const float* __restrict__ y, const float* __restrict__ g,
                             const float* __restrict__ be, float* __restrict__ scale,
                             float* __restrict__ bias, float eps) {
    int c = blockIdx.x;
    __shared__ float ssum[256], ssq[256];
    float s = 0.f, q = 0.f;
    for (int b = 0; b < BATCH; b++) {
        const float* p = y + ((size_t)(b * 64 + c)) * NYg * NXg;
        for (int i = threadIdx.x; i < NYg * NXg; i += 256) {
            float v = p[i]; s += v; q += v * v;
        }
    }
    ssum[threadIdx.x] = s; ssq[threadIdx.x] = q;
    __syncthreads();
    for (int off = 128; off >= 1; off >>= 1) {
        if (threadIdx.x < off) {
            ssum[threadIdx.x] += ssum[threadIdx.x + off];
            ssq[threadIdx.x]  += ssq[threadIdx.x + off];
        }
        __syncthreads();
    }
    if (threadIdx.x == 0) {
        float N = (float)(BATCH * NYg * NXg);
        float mean = ssum[0] / N;
        float var = ssq[0] / N - mean * mean;
        float rs = rsqrtf(var + eps);
        scale[c] = g[c] * rs;
        bias[c] = be[c] - mean * g[c] * rs;
    }
}

__global__ void k_bnrelu(const float* __restrict__ y, const float* __restrict__ scale,
                         const float* __restrict__ bias, float* __restrict__ x32,
                         _Float16* __restrict__ xh, int n) {
    int idx = blockIdx.x * 256 + threadIdx.x;
    if (idx >= n) return;
    int c = (idx / (NYg * NXg)) & 63;
    float v = fmaxf(0.f, y[idx] * scale[c] + bias[c]);
    x32[idx] = v;
    xh[idx] = (_Float16)v;
}

__global__ void k_head(const float* __restrict__ x32, const float* __restrict__ hw,
                       const float* __restrict__ hb, float* __restrict__ out) {
    int idx = blockIdx.x * 256 + threadIdx.x;
    if (idx >= BATCH * NYg * NXg) return;
    int b = idx / (NYg * NXg);
    int pix = idx - b * (NYg * NXg);
    const float* xp = x32 + ((size_t)b * 64) * NYg * NXg + pix;
    float a0 = hb[0], a1 = hb[1], a2 = hb[2];
    for (int i = 0; i < 64; i++) {
        float v = xp[(size_t)i * NYg * NXg];
        a0 += v * hw[i]; a1 += v * hw[64 + i]; a2 += v * hw[128 + i];
    }
    out[(size_t)idx * 3 + 0] = a0;
    out[(size_t)idx * 3 + 1] = a1;
    out[(size_t)idx * 3 + 2] = a2;
}

// ---------------------------------------------------------------- launch
extern "C" void kernel_launch(void* const* d_in, const int* in_sizes, int n_in,
                              void* d_out, int out_size, void* d_ws, size_t ws_size,
                              hipStream_t stream) {
    const float* points = (const float*)d_in[0];
    const float* pfn_w1 = (const float*)d_in[1];
    const float* pfn_g1 = (const float*)d_in[2];
    const float* pfn_b1 = (const float*)d_in[3];
    const float* pfn_w2 = (const float*)d_in[4];
    const float* pfn_g2 = (const float*)d_in[5];
    const float* pfn_b2 = (const float*)d_in[6];
    const float* pfn_w3 = (const float*)d_in[7];
    const float* pfn_g3 = (const float*)d_in[8];
    const float* pfn_b3 = (const float*)d_in[9];
    const float* cs_w  = (const float*)d_in[10];
    const float* cs_b  = (const float*)d_in[11];
    const float* cs_g  = (const float*)d_in[12];
    const float* cs_be = (const float*)d_in[13];
    const float* rpn_w  = (const float*)d_in[14];
    const float* rpn_b  = (const float*)d_in[15];
    const float* rpn_g  = (const float*)d_in[16];
    const float* rpn_be = (const float*)d_in[17];
    const float* head_w = (const float*)d_in[18];
    const float* head_b = (const float*)d_in[19];

    char* base = (char*)d_ws;
    size_t off = 0;
    auto alloc = [&](size_t bytes) -> void* {
        void* p = base + off;
        off += (bytes + 255) / 256 * 256;
        return p;
    };

    const size_t HW = (size_t)NYg * NXg;
    const size_t BCHW = (size_t)BATCH * 64 * HW;

    int*   cellcount = (int*)  alloc((size_t)BATCH * NCELL * 4);
    int*   ordinal   = (int*)  alloc((size_t)BATCH * NCELL * 4);
    int*   slot      = (int*)  alloc((size_t)BATCH * MAXV * 4);
    int*   vvv       = (int*)  alloc((size_t)BATCH * MAXV * 4);
    float* vox       = (float*)alloc((size_t)BATCH * MAXV * PV * 4 * 4);
    float* feat      = (float*)alloc((size_t)BATCH * MAXV * PV * 7 * 4);
    float* vmax1     = (float*)alloc((size_t)BATCH * MAXV * 32 * 4);
    float* vmax2     = (float*)alloc((size_t)BATCH * MAXV * 64 * 4);
    _Float16* wq1    = (_Float16*)alloc(32 * 32 * 2);
    _Float16* wq2    = (_Float16*)alloc(64 * 64 * 2);
    _Float16* wq3    = (_Float16*)alloc(96 * 64 * 2);
    float* colsum    = (float*)alloc(64 * 4);
    float* colsumsq  = (float*)alloc(64 * 4);
    float* bnscale   = (float*)alloc(64 * 4);
    float* bnbias    = (float*)alloc(64 * 4);
    float* dense     = (float*)alloc(BCHW * 4);
    _Float16* xh     = (_Float16*)alloc(BCHW * 2);
    float* y32       = (float*)alloc(BCHW * 4);
    float* x32       = (float*)alloc(BCHW * 4);
    _Float16* wc     = (_Float16*)alloc((size_t)5 * 64 * 576 * 2);
    float* cscale    = (float*)alloc(64 * 4);
    float* cbias     = (float*)alloc(64 * 4);

    // zero state we rely on
    hipMemsetAsync(cellcount, 0, (size_t)BATCH * NCELL * 4, stream);
    hipMemsetAsync(slot, 0, (size_t)BATCH * MAXV * 4, stream);
    hipMemsetAsync(vvv, 0, (size_t)BATCH * MAXV * 4, stream);
    hipMemsetAsync(vox, 0, (size_t)BATCH * MAXV * PV * 4 * 4, stream);
    hipMemsetAsync(dense, 0, BCHW * 4, stream);

    // voxelize
    k_count<<<(BATCH * NPTS + 255) / 256, 256, 0, stream>>>(points, cellcount);
    k_scan<<<BATCH, 1024, 0, stream>>>(cellcount, ordinal);
    k_fill<<<(BATCH * NPTS + 255) / 256, 256, 0, stream>>>(points, cellcount, ordinal, slot, vvv, vox);
    k_feat<<<BATCH * MAXV / 4, 128, 0, stream>>>(vox, slot, feat);

    // stage weights: PFN -> fragment-major B layout, conv -> fragment-major A layout
    k_pack_wB<<<(32 * 32 + 255) / 256, 256, 0, stream>>>(pfn_w1, wq1, 7, 32, 32);
    k_pack_wB<<<(64 * 64 + 255) / 256, 256, 0, stream>>>(pfn_w2, wq2, 39, 64, 64);
    k_pack_wB<<<(96 * 64 + 255) / 256, 256, 0, stream>>>(pfn_w3, wq3, 71, 96, 64);
    k_pack_convA<<<(2 * 64 * 576 + 255) / 256, 256, 0, stream>>>(cs_w, wc, 2 * 64 * 576);
    k_pack_convA<<<(3 * 64 * 576 + 255) / 256, 256, 0, stream>>>(rpn_w, wc + 2 * 64 * 576, 3 * 64 * 576);

    const float Nrows = (float)((size_t)BATCH * MAXV * PV);
    const int PFNG = BATCH * MAXV / 2;

    // PFN layer 1: K=7 (pad 32) -> 32
    hipMemsetAsync(colsum, 0, 64 * 4, stream);
    hipMemsetAsync(colsumsq, 0, 64 * 4, stream);
    k_pfn<<<PFNG, 128, 0, stream>>>(feat, nullptr, 0, 32, wq1, 32, 0, slot, vvv,
                                    colsum, colsumsq, nullptr, nullptr, nullptr, nullptr);
    k_finalize_bn<<<1, 64, 0, stream>>>(colsum, colsumsq, pfn_g1, pfn_b1, bnscale, bnbias, 32, Nrows, 1e-3f);
    k_pfn<<<PFNG, 128, 0, stream>>>(feat, nullptr, 0, 32, wq1, 32, 1, slot, vvv,
                                    nullptr, nullptr, bnscale, bnbias, vmax1, nullptr);

    // PFN layer 2: K=39 (pad 64) -> 64
    hipMemsetAsync(colsum, 0, 64 * 4, stream);
    hipMemsetAsync(colsumsq, 0, 64 * 4, stream);
    k_pfn<<<PFNG, 128, 0, stream>>>(feat, vmax1, 32, 64, wq2, 64, 0, slot, vvv,
                                    colsum, colsumsq, nullptr, nullptr, nullptr, nullptr);
    k_finalize_bn<<<1, 64, 0, stream>>>(colsum, colsumsq, pfn_g2, pfn_b2, bnscale, bnbias, 64, Nrows, 1e-3f);
    k_pfn<<<PFNG, 128, 0, stream>>>(feat, vmax1, 32, 64, wq2, 64, 1, slot, vvv,
                                    nullptr, nullptr, bnscale, bnbias, vmax2, nullptr);

    // PFN layer 3: K=71 (pad 96) -> 64, scatter-max into dense BEV
    hipMemsetAsync(colsum, 0, 64 * 4, stream);
    hipMemsetAsync(colsumsq, 0, 64 * 4, stream);
    k_pfn<<<PFNG, 128, 0, stream>>>(feat, vmax2, 64, 96, wq3, 64, 0, slot, vvv,
                                    colsum, colsumsq, nullptr, nullptr, nullptr, nullptr);
    k_finalize_bn<<<1, 64, 0, stream>>>(colsum, colsumsq, pfn_g3, pfn_b3, bnscale, bnbias, 64, Nrows, 1e-3f);
    k_pfn<<<PFNG, 128, 0, stream>>>(feat, vmax2, 64, 96, wq3, 64, 2, slot, vvv,
                                    nullptr, nullptr, bnscale, bnbias, nullptr, dense);

    // conv stack
    k_f32_to_f16<<<(int)((BCHW + 255) / 256), 256, 0, stream>>>(dense, xh, (int)BCHW);
    dim3 cgrid((NXg + 31) / 32, NYg, BATCH);
    for (int l = 0; l < 5; l++) {
        const _Float16* wl = wc + (size_t)l * 64 * 576;
        const float* bl = (l < 2) ? (cs_b + l * 64) : (rpn_b + (l - 2) * 64);
        const float* gl = (l < 2) ? (cs_g + l * 64) : (rpn_g + (l - 2) * 64);
        const float* el = (l < 2) ? (cs_be + l * 64) : (rpn_be + (l - 2) * 64);
        k_conv<<<cgrid, 256, 0, stream>>>(xh, wl, bl, y32);
        k_conv_stats<<<64, 256, 0, stream>>>(y32, gl, el, cscale, cbias, 1e-5f);
        k_bnrelu<<<(int)((BCHW + 255) / 256), 256, 0, stream>>>(y32, cscale, cbias, x32, xh, (int)BCHW);
    }
    k_head<<<(BATCH * NYg * NXg + 255) / 256, 256, 0, stream>>>(x32, head_w, head_b, (float*)d_out);
}
